// GCN_d_13116830122716
// MI455X (gfx1250) — matrix operand order repaired and checked by reading, not compiled, not run on hardware
//
#include <hip/hip_runtime.h>
#include <hip/hip_bf16.h>

typedef __attribute__((ext_vector_type(16))) _Float16 v16h;
typedef __attribute__((ext_vector_type(8)))  _Float16 v8h;
typedef __attribute__((ext_vector_type(8)))  float    v8f;

#define KNN      20
#define NPTS     1024
#define NB       8
#define M_NODES  (NB * NPTS)
#define E_KNN    (M_NODES * KNN)
#define BN_EPS   1e-5f
#define NT_PER_WAVE     4   // 4 N-tiles (64 cols) per wave -> 4 indep. WMMAs/K-step
#define WAVES_PER_BLOCK 4   // 4 M-tiles per block

// ---------------------------------------------------------------- utilities
__global__ void zero_i32(int* p, int n) {
    int i = blockIdx.x * blockDim.x + threadIdx.x;
    if (i < n) p[i] = 0;
}
__global__ void zero_f32(float* p, int n) {
    int i = blockIdx.x * blockDim.x + threadIdx.x;
    if (i < n) p[i] = 0.f;
}
__global__ void copy_i32(const int* __restrict__ s, int* __restrict__ d, int n) {
    int i = blockIdx.x * blockDim.x + threadIdx.x;
    if (i < n) d[i] = s[i];
}

// ---------------------------------------------------------------- kNN graph
__global__ __launch_bounds__(256) void knn_kernel(const float* __restrict__ x,
                                                  int* __restrict__ idx) {
    __shared__ float px[NPTS], py[NPTS], pz[NPTS], pxx[NPTS];
    int b = blockIdx.x;
    const float* xb = x + (size_t)b * 3 * NPTS;
    for (int n = threadIdx.x; n < NPTS; n += blockDim.x) {
        float a0 = xb[n], a1 = xb[NPTS + n], a2 = xb[2 * NPTS + n];
        px[n] = a0; py[n] = a1; pz[n] = a2;
        pxx[n] = a0 * a0 + a1 * a1 + a2 * a2;
    }
    __syncthreads();
    for (int i = threadIdx.x; i < NPTS; i += blockDim.x) {
        float xi = px[i], yi = py[i], zi = pz[i], xxi = pxx[i];
        float bestv[KNN];
        int   bestj[KNN];
#pragma unroll
        for (int k = 0; k < KNN; k++) { bestv[k] = -3.4e38f; bestj[k] = 0; }
        for (int j = 0; j < NPTS; j++) {
            float inner = xi * px[j] + yi * py[j] + zi * pz[j];
            float v = 2.0f * inner - xxi - pxx[j];   // -||xi - xj||^2
            if (v > bestv[KNN - 1]) {
                int p = KNN - 1;
                while (p > 0 && bestv[p - 1] < v) {  // ties keep earlier index
                    bestv[p] = bestv[p - 1]; bestj[p] = bestj[p - 1]; --p;
                }
                bestv[p] = v; bestj[p] = j;
            }
        }
        int* out = idx + (size_t)(b * NPTS + i) * KNN;
#pragma unroll
        for (int k = 0; k < KNN; k++) out[k] = b * NPTS + bestj[k];
    }
}

__global__ void count_kernel(const int* __restrict__ idx, int* indeg, int E) {
    int e = blockIdx.x * blockDim.x + threadIdx.x;
    if (e < E) atomicAdd(&indeg[idx[e]], 1);
}

// Single-block exclusive scan over 8192 ints (1024 threads x 8 elems, LDS scan)
__global__ __launch_bounds__(1024) void scan_kernel(const int* __restrict__ indeg,
                                                    int* __restrict__ rowptr) {
    __shared__ int s[1024];
    int tid  = threadIdx.x;
    int base = tid * 8;
    int local[8];
    int sum = 0;
#pragma unroll
    for (int i = 0; i < 8; i++) { local[i] = sum; sum += indeg[base + i]; }
    s[tid] = sum;
    __syncthreads();
    for (int off = 1; off < 1024; off <<= 1) {
        int v = (tid >= off) ? s[tid - off] : 0;
        __syncthreads();
        s[tid] += v;
        __syncthreads();
    }
    int pre = (tid == 0) ? 0 : s[tid - 1];
#pragma unroll
    for (int i = 0; i < 8; i++) rowptr[base + i] = pre + local[i];
    if (tid == 1023) rowptr[M_NODES] = s[1023];
}

__global__ void fill_kernel(const int* __restrict__ idx, int* cursor,
                            int* __restrict__ colsrc, int E) {
    int e = blockIdx.x * blockDim.x + threadIdx.x;
    if (e >= E) return;
    int i = e / KNN;            // source (center) node
    int j = idx[e];             // destination node
    int pos = atomicAdd(&cursor[j], 1);
    colsrc[pos] = i;
}

__global__ void dinv_kernel(const int* __restrict__ indeg, float* __restrict__ dinv, int n) {
    int i = blockIdx.x * blockDim.x + threadIdx.x;
    if (i < n) dinv[i] = rsqrtf((float)(indeg[i] + 1));   // +1 self loop
}

// ---------------------------------------------------------------- f16 prep
// x [B,3,N] -> node features [M, 32] f16 (K padded 3 -> 32 with zeros)
__global__ void prep_input(const float* __restrict__ x, _Float16* __restrict__ h) {
    int i = blockIdx.x * blockDim.x + threadIdx.x;
    if (i >= M_NODES * 32) return;
    int row = i >> 5, c = i & 31;
    int b = row >> 10, p = row & 1023;
    float v = (c < 3) ? x[((size_t)b * 3 + c) * NPTS + p] : 0.f;
    h[i] = (_Float16)v;
}

// W [Kc, Nc] f32 -> Wt [Np, Kp] f16 (transposed, zero padded; Np mult of 64)
__global__ void prep_weight(const float* __restrict__ W, _Float16* __restrict__ Wt,
                            int Kc, int Nc, int Kp, int Np) {
    int k = blockIdx.x * blockDim.x + threadIdx.x;
    int n = blockIdx.y;
    if (k >= Kp || n >= Np) return;
    float v = (n < Nc && k < Kc) ? W[(size_t)k * Nc + n] : 0.f;
    Wt[(size_t)n * Kp + k] = (_Float16)v;
}

// ---------------------------------------------------------------- WMMA GEMM
// C[M,Nc] = A[M,Kp](f16) * Wt[Np,Kp]^T(f16)  (+bias), f32 accumulate.
// One wave: 1 M-tile x 4 N-tiles. Per K-step: all 4 B fragments are loaded
// into distinct registers first, then 4 independent WMMAs issue back-to-back,
// so VMEM waits can be partial and overlap the XDL pipe.
__global__ __launch_bounds__(WAVES_PER_BLOCK * 32) void wmma_gemm(
    const _Float16* __restrict__ A, const _Float16* __restrict__ Bt,
    const float* __restrict__ bias, float* __restrict__ C,
    int M, int Nc, int Kp) {
    int lane   = threadIdx.x & 31;
    int wave   = threadIdx.x >> 5;
    int mtiles = (M + 15) >> 4;
    int tileM  = blockIdx.x * WAVES_PER_BLOCK + wave;
    if (tileM >= mtiles) return;          // whole-wave exit: EXEC stays full
    int g  = lane >> 4;                   // lane group (K-half select)
    int lm = lane & 15;
    int arow = tileM * 16 + lm;
    if (arow >= M) arow = M - 1;          // clamp instead of guard
    int n0 = blockIdx.y * (NT_PER_WAVE * 16);
    const _Float16* ap  = A  + (size_t)arow * Kp;
    const _Float16* bp0 = Bt + (size_t)(n0 + lm) * Kp;
    v8f zero = {};
    v8f acc[NT_PER_WAVE] = {zero, zero, zero, zero};
    for (int kb = 0; kb < Kp; kb += 32) {
        // A fragment (reused by all 4 WMMAs)
        v8h alo = *(const v8h*)(ap + kb + g * 8);        // K = kb + g*8 .. +7
        v8h ahi = *(const v8h*)(ap + kb + 16 + g * 8);   // K = kb + 16 + g*8 ..
        v16h af;
#pragma unroll
        for (int e = 0; e < 8; e++) { af[e] = alo[e]; af[e + 8] = ahi[e]; }
        int pk = (kb + 32 < Kp) ? kb + 32 : kb;          // branchless prefetch
        __builtin_prefetch(ap + pk, 0, 1);               // global_prefetch_b8
        // Phase 1: issue all 4 B-fragment loads (distinct live ranges)
        v16h bf[NT_PER_WAVE];
#pragma unroll
        for (int nt = 0; nt < NT_PER_WAVE; nt++) {
            const _Float16* bp = bp0 + (size_t)nt * 16 * Kp;
            v8h blo = *(const v8h*)(bp + kb + g * 8);    // K = kb + g*8 .. +7
            v8h bhi = *(const v8h*)(bp + kb + 16 + g * 8);
#pragma unroll
            for (int e = 0; e < 8; e++) { bf[nt][e] = blo[e]; bf[nt][e + 8] = bhi[e]; }
        }
        // Phase 2: 4 independent back-to-back WMMAs sharing the A fragment
#pragma unroll
        for (int nt = 0; nt < NT_PER_WAVE; nt++) {
            acc[nt] = __builtin_amdgcn_wmma_f32_16x16x32_f16(
                false, af, false, bf[nt], (short)0, acc[nt], false, false);
        }
    }
#pragma unroll
    for (int nt = 0; nt < NT_PER_WAVE; nt++) {
        int col = n0 + nt * 16 + lm;
        if (col < Nc) {
            float bv = bias ? bias[col] : 0.f;
#pragma unroll
            for (int v = 0; v < 8; v++) {
                int r = tileM * 16 + g * 8 + v;          // C: row = g*8+v, col = lm
                if (r < M) C[(size_t)r * Nc + col] = acc[nt][v] + bv;
            }
        }
    }
}

// ---------------------------------------------------------------- aggregation
// out[j,c] = dinv[j] * ( dinv[j]*t[j,c] + sum_in dinv[i]*t[i,c] ) + bias[c]
__global__ void aggregate(const float* __restrict__ t, const int* __restrict__ rowptr,
                          const int* __restrict__ colsrc, const float* __restrict__ dinv,
                          const float* __restrict__ bias, float* __restrict__ out, int d) {
    int j = blockIdx.x;
    int c = blockIdx.y * blockDim.x + threadIdx.x;
    if (c >= d) return;
    float dj  = dinv[j];
    float acc = dj * t[(size_t)j * d + c];
    int s = rowptr[j], e = rowptr[j + 1];
    for (int k = s; k < e; k++) {
        int i = colsrc[k];
        acc += dinv[i] * t[(size_t)i * d + c];
    }
    out[(size_t)j * d + c] = dj * acc + bias[c];
}

// ---------------------------------------------------------------- batch norm
__global__ __launch_bounds__(256) void bn_stats(const float* __restrict__ a,
                                                float* sums, float* sumsq, int d) {
    int r0 = blockIdx.x * 64;
    float s0[4] = {0, 0, 0, 0}, s1[4] = {0, 0, 0, 0};
    for (int r = r0; r < r0 + 64; r++) {
        int ci = 0;
        for (int c = threadIdx.x; c < d; c += 256, ci++) {
            float v = a[(size_t)r * d + c];
            s0[ci] += v; s1[ci] += v * v;
        }
    }
    int ci = 0;
    for (int c = threadIdx.x; c < d; c += 256, ci++) {
        atomicAdd(&sums[c],  s0[ci]);
        atomicAdd(&sumsq[c], s1[ci]);
    }
}

__global__ void bn_finalize(const float* __restrict__ sums, const float* __restrict__ sumsq,
                            const float* __restrict__ g, const float* __restrict__ bt,
                            float* scale, float* shift, int d, float invM) {
    int c = blockIdx.x * blockDim.x + threadIdx.x;
    if (c >= d) return;
    float m  = sums[c] * invM;
    float v  = sumsq[c] * invM - m * m;       // biased variance
    float sc = g[c] * rsqrtf(v + BN_EPS);
    scale[c] = sc;
    shift[c] = bt[c] - m * sc;
}

__global__ void bn_apply(const float* __restrict__ a, const float* __restrict__ scale,
                         const float* __restrict__ shift, _Float16* __restrict__ h16,
                         float* __restrict__ h32, int d, int total) {
    int i = blockIdx.x * blockDim.x + threadIdx.x;
    if (i >= total) return;
    int c = i % d;
    float v = a[i] * scale[c] + shift[c];
    v = (v >= 0.f) ? v : 0.2f * v;            // leaky relu 0.2
    h16[i] = (_Float16)v;
    if (h32) h32[i] = v;
}

// ---------------------------------------------------------------- pooling
__global__ void pool_kernel(const float* __restrict__ h, float* __restrict__ pooled,
                            _Float16* __restrict__ pooled16, int d) {
    int b = blockIdx.x;
    int c = blockIdx.y * blockDim.x + threadIdx.x;
    if (c >= d) return;
    float s = 0.f;
    for (int n = 0; n < NPTS; n++) s += h[((size_t)b * NPTS + n) * d + c];
    float mean = s * (1.0f / NPTS);
    pooled  [(size_t)b * 2 * d + c]     = mean;
    pooled  [(size_t)b * 2 * d + d + c] = s;
    pooled16[(size_t)b * 2 * d + c]     = (_Float16)mean;
    pooled16[(size_t)b * 2 * d + d + c] = (_Float16)s;
}

// BN over 8 rows + leaky relu, fused (FC layers)
__global__ void fc_bn_lrelu(const float* __restrict__ a, const float* __restrict__ g,
                            const float* __restrict__ bt, _Float16* __restrict__ h16,
                            int rows, int d) {
    int c = blockIdx.x * blockDim.x + threadIdx.x;
    if (c >= d) return;
    float m = 0.f;
    for (int r = 0; r < rows; r++) m += a[(size_t)r * d + c];
    m /= rows;
    float v = 0.f;
    for (int r = 0; r < rows; r++) { float t = a[(size_t)r * d + c] - m; v += t * t; }
    v /= rows;
    float sc = g[c] * rsqrtf(v + BN_EPS);
    for (int r = 0; r < rows; r++) {
        float t = (a[(size_t)r * d + c] - m) * sc + bt[c];
        t = (t >= 0.f) ? t : 0.2f * t;
        h16[(size_t)r * d + c] = (_Float16)t;
    }
}

// ---------------------------------------------------------------- launch
extern "C" void kernel_launch(void* const* d_in, const int* in_sizes, int n_in,
                              void* d_out, int out_size, void* d_ws, size_t ws_size,
                              hipStream_t stream) {
    const float* x = (const float*)d_in[0];
    const float* Wl1 = (const float*)d_in[21];
    const float* g6  = (const float*)d_in[22];
    const float* bt6 = (const float*)d_in[23];
    const float* Wl2 = (const float*)d_in[24];
    const float* bl2 = (const float*)d_in[25];
    const float* g7  = (const float*)d_in[26];
    const float* bt7 = (const float*)d_in[27];
    const float* Wl3 = (const float*)d_in[28];
    const float* bl3 = (const float*)d_in[29];

    // workspace carve
    char* ws = (char*)d_ws;
    auto carve = [&](size_t bytes) -> char* {
        char* p = ws;
        ws += (bytes + 255) & ~(size_t)255;
        return p;
    };
    int*       idx     = (int*)      carve((size_t)E_KNN * 4);
    int*       colsrc  = (int*)      carve((size_t)E_KNN * 4);
    int*       indeg   = (int*)      carve((size_t)M_NODES * 4);
    int*       rowptr  = (int*)      carve((size_t)(M_NODES + 1) * 4);
    int*       cursor  = (int*)      carve((size_t)M_NODES * 4);
    float*     dinv    = (float*)    carve((size_t)M_NODES * 4);
    float*     t       = (float*)    carve((size_t)M_NODES * 1024 * 4);
    float*     agg     = (float*)    carve((size_t)M_NODES * 1024 * 4);
    _Float16*  h16a    = (_Float16*) carve((size_t)M_NODES * 1024 * 2);
    _Float16*  h16b    = (_Float16*) carve((size_t)M_NODES * 1024 * 2);
    _Float16*  Wt      = (_Float16*) carve((size_t)2048 * 512 * 2);
    float*     bnsum   = (float*)    carve(1024 * 4);
    float*     bnsq    = (float*)    carve(1024 * 4);
    float*     scale   = (float*)    carve(1024 * 4);
    float*     shift   = (float*)    carve(1024 * 4);
    float*     pooled  = (float*)    carve((size_t)NB * 2048 * 4);
    _Float16*  pooled16= (_Float16*) carve((size_t)NB * 2048 * 2);
    float*     fc1     = (float*)    carve((size_t)NB * 512 * 4);
    _Float16*  fc1h    = (_Float16*) carve((size_t)NB * 512 * 2);
    float*     fc2     = (float*)    carve((size_t)NB * 256 * 4);
    _Float16*  fc2h    = (_Float16*) carve((size_t)NB * 256 * 2);

    const int M = M_NODES;
    // ---- graph build (fixed across all layers)
    knn_kernel<<<NB, 256, 0, stream>>>(x, idx);
    zero_i32<<<(M + 255) / 256, 256, 0, stream>>>(indeg, M);
    count_kernel<<<(E_KNN + 255) / 256, 256, 0, stream>>>(idx, indeg, E_KNN);
    scan_kernel<<<1, 1024, 0, stream>>>(indeg, rowptr);
    copy_i32<<<(M + 255) / 256, 256, 0, stream>>>(rowptr, cursor, M);
    fill_kernel<<<(E_KNN + 255) / 256, 256, 0, stream>>>(idx, cursor, colsrc, E_KNN);
    dinv_kernel<<<(M + 255) / 256, 256, 0, stream>>>(indeg, dinv, M);
    prep_input<<<(M * 32 + 255) / 256, 256, 0, stream>>>(x, h16a);

    // ---- 5 GCN layers
    const int dims[6] = {3, 64, 128, 256, 512, 1024};
    _Float16* hin = h16a;
    _Float16* hout = h16b;
    for (int l = 1; l <= 5; l++) {
        int din  = dims[l - 1];
        int dout = dims[l];
        int Kp   = (l == 1) ? 32 : din;       // K padded to 32 for layer 1
        const float* W  = (const float*)d_in[1 + (l - 1) * 4 + 0];
        const float* bb = (const float*)d_in[1 + (l - 1) * 4 + 1];
        const float* gg = (const float*)d_in[1 + (l - 1) * 4 + 2];
        const float* bt = (const float*)d_in[1 + (l - 1) * 4 + 3];
        int Np     = ((dout + 63) / 64) * 64; // pad cols to wave N-span
        prep_weight<<<dim3((Kp + 255) / 256, Np), 256, 0, stream>>>(W, Wt, din, dout, Kp, Np);
        wmma_gemm<<<dim3(M / 16 / WAVES_PER_BLOCK, Np / 64),
                    WAVES_PER_BLOCK * 32, 0, stream>>>(hin, Wt, nullptr, t, M, dout, Kp);
        aggregate<<<dim3(M, (dout + 255) / 256), 256, 0, stream>>>(
            t, rowptr, colsrc, dinv, bb, agg, dout);
        zero_f32<<<(dout + 255) / 256, 256, 0, stream>>>(bnsum, dout);
        zero_f32<<<(dout + 255) / 256, 256, 0, stream>>>(bnsq, dout);
        bn_stats<<<M / 64, 256, 0, stream>>>(agg, bnsum, bnsq, dout);
        bn_finalize<<<(dout + 255) / 256, 256, 0, stream>>>(
            bnsum, bnsq, gg, bt, scale, shift, dout, 1.0f / M);
        float* h32out = (l == 5) ? t : nullptr;   // reuse t for final f32 features
        bn_apply<<<(M * dout + 255) / 256, 256, 0, stream>>>(
            agg, scale, shift, hout, h32out, dout, M * dout);
        _Float16* tmp = hin; hin = hout; hout = tmp;
    }

    // ---- global mean+sum pooling -> [8, 2048]
    pool_kernel<<<dim3(NB, (1024 + 255) / 256), 256, 0, stream>>>(t, pooled, pooled16, 1024);

    // ---- FC1: [8,2048] x [2048,512], BN, lrelu
    prep_weight<<<dim3((2048 + 255) / 256, 512), 256, 0, stream>>>(Wl1, Wt, 2048, 512, 2048, 512);
    wmma_gemm<<<dim3(1, 512 / 64), WAVES_PER_BLOCK * 32, 0, stream>>>(
        pooled16, Wt, nullptr, fc1, NB, 512, 2048);
    fc_bn_lrelu<<<(512 + 255) / 256, 256, 0, stream>>>(fc1, g6, bt6, fc1h, NB, 512);

    // ---- FC2: [8,512] x [512,256] + bias, BN, lrelu
    prep_weight<<<dim3((512 + 255) / 256, 256), 256, 0, stream>>>(Wl2, Wt, 512, 256, 512, 256);
    wmma_gemm<<<dim3(1, 256 / 64), WAVES_PER_BLOCK * 32, 0, stream>>>(
        fc1h, Wt, bl2, fc2, NB, 256, 512);
    fc_bn_lrelu<<<1, 256, 0, stream>>>(fc2, g7, bt7, fc2h, NB, 256);

    // ---- FC3: [8,256] x [256,40] + bias -> d_out (Wt padded to 64 cols)
    prep_weight<<<dim3(1, 64), 256, 0, stream>>>(Wl3, Wt, 256, 40, 256, 64);
    wmma_gemm<<<dim3(1, 1), WAVES_PER_BLOCK * 32, 0, stream>>>(
        fc2h, Wt, bl3, (float*)d_out, NB, 40, 256);
}
